// LinearAttention_62998580298120
// MI455X (gfx1250) — matrix operand order repaired and checked by reading, not compiled, not run on hardware
//
#include <hip/hip_runtime.h>
#include <hip/hip_bf16.h>
#include <cstdint>

// ---------------------------------------------------------------------------
// CDNA5 (gfx1250) wave32 WMMA implementation of the gated causal
// linear-attention block. All matmuls use v_wmma_f32_16x16x32_bf16
// (bf16 inputs, f32 accumulate). fp32 inputs are converted to bf16 once.
// Straight global->LDS tile copies use GLOBAL_LOAD_ASYNC_TO_LDS_B128
// (ASYNCcnt path); transposing copies batch loads before LDS stores.
// ---------------------------------------------------------------------------

typedef __bf16 bf16_t;
typedef __attribute__((ext_vector_type(16))) __bf16 v16bf;
typedef __attribute__((ext_vector_type(8)))  __bf16 v8bf;
typedef __attribute__((ext_vector_type(8)))  float  v8f;

#define N_TOK  2048
#define EMBED  1024
#define HEADS  16
#define HD     64

union cvt_u32 { unsigned u; bf16_t b[2]; };

__device__ __forceinline__ v16bf cat8(v8bf a, v8bf b) {
  return __builtin_shufflevector(a, b, 0, 1, 2, 3, 4, 5, 6, 7,
                                 8, 9, 10, 11, 12, 13, 14, 15);
}

// Async 16-byte global -> LDS copy (gfx1250 VGLOBAL async op, ASYNCcnt).
// VDST holds the workgroup-relative LDS byte offset = low 32 bits of the
// flat __shared__ pointer (ISA 10.2: LDS aperture keeps offset in addr[31:0]).
__device__ __forceinline__ void async_copy_b128(void* lds, const void* g) {
  asm volatile("global_load_async_to_lds_b128 %0, %1, off"
               :: "v"((unsigned)(uintptr_t)lds),
                  "v"((unsigned long long)(uintptr_t)g)
               : "memory");
}
__device__ __forceinline__ void wait_async_lds() {
  asm volatile("s_wait_asynccnt 0x0" ::: "memory");
}

// ---------------- WMMA fragment helpers (contiguous k order for A and B) ----------------
// A operand 16x32 (MxK), source row-major [m][k]:
//   lane<16 : row=lane,    K {0..15}
//   lane>=16: row=lane-16, K {16..31}
__device__ __forceinline__ v16bf frag_a_row(const bf16_t* s, int ld, int rowBase, int kOff) {
  const int lane = threadIdx.x & 31;
  const int hi   = lane >> 4;
  const bf16_t* p = s + (rowBase + (lane & 15)) * ld + kOff + 16 * hi;
  v8bf lo  = *(const v8bf*)p;
  v8bf hi8 = *(const v8bf*)(p + 8);
  return cat8(lo, hi8);
}

// B operand 32x16 (KxN): per-lane column n = lane&15, K = 2j + 16*(lane>>4).
// Source stored [n][k] (i.e. B^T row-major) -> 16 contiguous bf16 per lane.
__device__ __forceinline__ v16bf frag_b_nk(const bf16_t* s, int ld, int colBase, int kOff) {
  const int lane = threadIdx.x & 31;
  const bf16_t* p = s + (colBase + (lane & 15)) * ld + kOff + 16 * (lane >> 4);
  v8bf lo  = *(const v8bf*)p;
  v8bf hi8 = *(const v8bf*)(p + 8);
  return cat8(lo, hi8);
}

__device__ __forceinline__ v8f wmma_bf16(v16bf a, v16bf b, v8f c) {
  return __builtin_amdgcn_wmma_f32_16x16x32_bf16(false, a, false, b, (short)0, c, false, false);
}

// ---------------------------- fp32 -> bf16 convert ---------------------------
__global__ __launch_bounds__(256) void cvt_f32_bf16(const float* __restrict__ src,
                                                    bf16_t* __restrict__ dst, int n) {
  int i = blockIdx.x * 256 + threadIdx.x;
  if (i < n) dst[i] = (bf16_t)src[i];
}

// --------------------------------- GEMM -------------------------------------
// C[M,N] = A[M,K] @ B[K,N]  (bf16 in, f32 accumulate). epi: 0=none, 1=silu.
// Block 256 threads = 8 waves; tile 128x128x32; each wave computes 32x64.
// GUARD=false: N assumed a multiple of BN (no bounds checks anywhere).
#define BM 128
#define BN 128
#define BK 32
#define ALD 40    // 32 + 8 pad; 80-byte rows keep 16B chunk alignment
#define BTLD 40   // Bt is [n][k]: 128 rows of 32(+8 pad)

template <bool GUARD>
__global__ __launch_bounds__(256) void gemm_bf16_kernel(
    const bf16_t* __restrict__ A, const bf16_t* __restrict__ B,
    float* __restrict__ Cf, bf16_t* __restrict__ Cb,
    int M, int N, int K, int epi) {
  __shared__ __align__(16) bf16_t As[BM * ALD];
  __shared__ __align__(16) bf16_t Bt[BN * BTLD];   // transposed B tile [n][k]

  const int tid   = threadIdx.x;
  const int mBase = blockIdx.y * BM;
  const int nBase = blockIdx.x * BN;
  const int wave  = tid >> 5;
  const int wm    = wave & 3;   // 4 row-groups of 32
  const int wn    = wave >> 2;  // 2 col-groups of 64
  const int lane  = tid & 31;
  const int hi    = lane >> 4;
  const int ln    = lane & 15;

  v8f acc[2][4];
#pragma unroll
  for (int t = 0; t < 2; ++t)
#pragma unroll
    for (int c = 0; c < 4; ++c) acc[t][c] = (v8f)0.0f;

  const unsigned* B32 = (const unsigned*)B;
  const int N2 = N >> 1;

  for (int kBase = 0; kBase < K; kBase += BK) {
    // A tile: 128x32 bf16 = 512 x 16B chunks; 2 async copies per thread
#pragma unroll
    for (int i = 0; i < 2; ++i) {
      const int idx = tid * 2 + i;
      const int row = idx >> 2, c4 = idx & 3;
      async_copy_b128(&As[row * ALD + c4 * 8],
                      A + (size_t)(mBase + row) * K + kBase + c4 * 8);
    }
    // B tile: read [k][n] dwords coalesced (all issued first), then store
    // transposed into Bt[n][k]
    unsigned bv[8];
#pragma unroll
    for (int i = 0; i < 8; ++i) {
      const int idx = tid * 8 + i;
      const int row = idx >> 6, c2 = idx & 63;   // row = k (0..31), c2 = n-pair
      if (GUARD) {
        const int col = nBase + c2 * 2;
        bv[i] = (col < N) ? B32[(size_t)(kBase + row) * N2 + (col >> 1)] : 0u;
      } else {
        bv[i] = B32[(size_t)(kBase + row) * N2 + (nBase >> 1) + c2];
      }
    }
#pragma unroll
    for (int i = 0; i < 8; ++i) {
      const int idx = tid * 8 + i;
      const int row = idx >> 6, c2 = idx & 63;
      cvt_u32 cv; cv.u = bv[i];
      Bt[(c2 * 2)     * BTLD + row] = cv.b[0];
      Bt[(c2 * 2 + 1) * BTLD + row] = cv.b[1];
    }
    if (kBase + BK < K)   // prefetch next B tile -> global_prefetch_b8
      __builtin_prefetch(B + (size_t)(kBase + BK + (tid >> 3)) * N + nBase, 0, 1);
    wait_async_lds();
    __syncthreads();

    v16bf a0 = frag_a_row(As, ALD, wm * 32, 0);
    v16bf a1 = frag_a_row(As, ALD, wm * 32 + 16, 0);
#pragma unroll
    for (int c = 0; c < 4; ++c) {
      v16bf b = frag_b_nk(Bt, BTLD, wn * 64 + c * 16, 0);
      acc[0][c] = wmma_bf16(a0, b, acc[0][c]);
      acc[1][c] = wmma_bf16(a1, b, acc[1][c]);
    }
    __syncthreads();
  }

#pragma unroll
  for (int t = 0; t < 2; ++t)
#pragma unroll
    for (int c = 0; c < 4; ++c)
#pragma unroll
      for (int r = 0; r < 8; ++r) {
        const int row = mBase + wm * 32 + t * 16 + r + 8 * hi;
        const int col = nBase + wn * 64 + c * 16 + ln;
        if (!GUARD || col < N) {
          float v = acc[t][c][r];
          if (epi == 1) v = v / (1.0f + __expf(-v));  // SiLU
          if (Cf) Cf[(size_t)row * N + col] = v;
          if (Cb) Cb[(size_t)row * N + col] = (bf16_t)v;
        }
      }
}

// ---------------------------------- RoPE -------------------------------------
// In-place on bf16 [N_TOK, EMBED] viewed as heads of HD=64: pairs (i, i+32).
__global__ __launch_bounds__(256) void rope_kernel(bf16_t* __restrict__ t) {
  const int idx = blockIdx.x * 256 + threadIdx.x;  // N_TOK*HEADS*32 total
  const int i = idx & 31;
  const int h = (idx >> 5) & (HEADS - 1);
  const int n = idx >> 9;
  const float inv = __powf(10000.0f, -(float)i / 32.0f);
  float sn, cs;
  __sincosf((float)n * inv, &sn, &cs);
  const size_t base = (size_t)n * EMBED + h * HD;
  const float t1 = (float)t[base + i];
  const float t2 = (float)t[base + 32 + i];
  t[base + i]      = (bf16_t)(t1 * cs - t2 * sn);
  t[base + 32 + i] = (bf16_t)(t2 * cs + t1 * sn);
}

// ------------------------- causal masked attention ---------------------------
// One workgroup (4 waves) per (head, 64-query tile). out = (Q K^T .* tril) V.
// Qs, Ks: [token][dim] (A / B^T layouts, contiguous per-lane).
// Vt:     [dim][key]   (transposed -> contiguous B fragments for S @ V).
#define TLD 72   // 64 + 8 pad; 144-byte rows keep 16B alignment

__global__ __launch_bounds__(128) void attn_kernel(
    const bf16_t* __restrict__ Q, const bf16_t* __restrict__ Km,
    const bf16_t* __restrict__ V, float* __restrict__ O) {
  __shared__ __align__(16) bf16_t Qs[64 * TLD];
  __shared__ __align__(16) bf16_t Ks[64 * TLD];
  __shared__ __align__(16) bf16_t Vt[64 * TLD];
  __shared__ __align__(16) bf16_t Ss[64 * TLD];

  const int head = blockIdx.x;
  const int qt   = blockIdx.y;
  const int tid  = threadIdx.x;
  const int wave = tid >> 5;
  const int lane = tid & 31;
  const int hi   = lane >> 4;
  const int ln   = lane & 15;
  const int colOff = head * HD;

  const unsigned* V32 = (const unsigned*)V;
  const int E2 = EMBED >> 1;

  // Q tile: 64 rows x 8 16B chunks; async copy, 4 per thread
#pragma unroll
  for (int i = 0; i < 4; ++i) {
    const int idx = tid * 4 + i;
    const int row = idx >> 3, c4 = idx & 7;
    async_copy_b128(&Qs[row * TLD + c4 * 8],
                    Q + (size_t)(qt * 64 + row) * EMBED + colOff + c4 * 8);
  }
  wait_async_lds();
  __syncthreads();

  const v16bf qf0 = frag_a_row(Qs, TLD, wave * 16, 0);
  const v16bf qf1 = frag_a_row(Qs, TLD, wave * 16, 32);

  v8f oacc[4];
#pragma unroll
  for (int d = 0; d < 4; ++d) oacc[d] = (v8f)0.0f;

  for (int kt = 0; kt <= qt; ++kt) {
    __syncthreads();  // previous iteration fully consumed
    // K tile: straight async copy; V tile: batched loads then LDS transpose
#pragma unroll
    for (int i = 0; i < 4; ++i) {
      const int idx = tid * 4 + i;
      const int row = idx >> 3, c4 = idx & 7;
      async_copy_b128(&Ks[row * TLD + c4 * 8],
                      Km + (size_t)(kt * 64 + row) * EMBED + colOff + c4 * 8);
    }
    unsigned vv[16];
#pragma unroll
    for (int i = 0; i < 16; ++i) {
      const int idx = tid * 16 + i;
      const int row = idx >> 5, c2 = idx & 31;   // row = key, c2 = dim-pair
      vv[i] = V32[(size_t)(kt * 64 + row) * E2 + (colOff >> 1) + c2];
    }
#pragma unroll
    for (int i = 0; i < 16; ++i) {
      const int idx = tid * 16 + i;
      const int row = idx >> 5, c2 = idx & 31;
      cvt_u32 cv; cv.u = vv[i];
      Vt[(c2 * 2)     * TLD + row] = cv.b[0];
      Vt[(c2 * 2 + 1) * TLD + row] = cv.b[1];
    }
    wait_async_lds();
    __syncthreads();

    // S(16q x 64k) = Q_wave (16x64) @ K_tile^T ; causal mask; restage as bf16
#pragma unroll
    for (int c = 0; c < 4; ++c) {
      v16bf kb0 = frag_b_nk(Ks, TLD, c * 16, 0);
      v16bf kb1 = frag_b_nk(Ks, TLD, c * 16, 32);
      v8f s = (v8f)0.0f;
      s = wmma_bf16(qf0, kb0, s);
      s = wmma_bf16(qf1, kb1, s);
      if (kt == qt) {
#pragma unroll
        for (int r = 0; r < 8; ++r) {
          const int key  = c * 16 + ln;
          const int qrow = wave * 16 + r + 8 * hi;
          if (key > qrow) s[r] = 0.0f;   // energy * tril (no softmax)
        }
      }
#pragma unroll
      for (int r = 0; r < 8; ++r)
        Ss[(wave * 16 + r + 8 * hi) * TLD + c * 16 + ln] = (bf16_t)s[r];
    }
    __syncthreads();

    const v16bf sa0 = frag_a_row(Ss, TLD, wave * 16, 0);
    const v16bf sa1 = frag_a_row(Ss, TLD, wave * 16, 32);
#pragma unroll
    for (int d = 0; d < 4; ++d) {
      v16bf vb0 = frag_b_nk(Vt, TLD, d * 16, 0);
      v16bf vb1 = frag_b_nk(Vt, TLD, d * 16, 32);
      oacc[d] = wmma_bf16(sa0, vb0, oacc[d]);
      oacc[d] = wmma_bf16(sa1, vb1, oacc[d]);
    }
  }

#pragma unroll
  for (int d = 0; d < 4; ++d)
#pragma unroll
    for (int r = 0; r < 8; ++r) {
      const int qrow = qt * 64 + wave * 16 + r + 8 * hi;
      O[(size_t)qrow * EMBED + colOff + d * 16 + ln] = oacc[d][r];
    }
}

// ------------------------- gate * sigmoid + LayerNorm ------------------------
__global__ __launch_bounds__(256) void gate_ln_kernel(
    const float* __restrict__ attn, const float* __restrict__ gateLogit,
    const float* __restrict__ lnw, const float* __restrict__ lnb,
    bf16_t* __restrict__ out) {
  __shared__ float red[256];
  const int row = blockIdx.x;
  const int tid = threadIdx.x;
  float g[4];
  float sum = 0.f;
#pragma unroll
  for (int i = 0; i < 4; ++i) {
    const int c = tid + i * 256;
    const float a = attn[(size_t)row * EMBED + c];
    const float z = gateLogit[(size_t)row * EMBED + c];
    g[i] = a * (1.0f / (1.0f + __expf(-z)));
    sum += g[i];
  }
  red[tid] = sum;
  __syncthreads();
  for (int s = 128; s > 0; s >>= 1) {
    if (tid < s) red[tid] += red[tid + s];
    __syncthreads();
  }
  const float mu = red[0] * (1.0f / EMBED);
  __syncthreads();
  float vs = 0.f;
#pragma unroll
  for (int i = 0; i < 4; ++i) { const float d = g[i] - mu; vs += d * d; }
  red[tid] = vs;
  __syncthreads();
  for (int s = 128; s > 0; s >>= 1) {
    if (tid < s) red[tid] += red[tid + s];
    __syncthreads();
  }
  const float rs = rsqrtf(red[0] * (1.0f / EMBED) + 1e-5f);
#pragma unroll
  for (int i = 0; i < 4; ++i) {
    const int c = tid + i * 256;
    out[(size_t)row * EMBED + c] = (bf16_t)((g[i] - mu) * rs * lnw[c] + lnb[c]);
  }
}

// --------------------------------- launch ------------------------------------
extern "C" void kernel_launch(void* const* d_in, const int* in_sizes, int n_in,
                              void* d_out, int out_size, void* d_ws, size_t ws_size,
                              hipStream_t stream) {
  const float* x   = (const float*)d_in[0];
  const float* Wq  = (const float*)d_in[1];
  const float* Wk  = (const float*)d_in[2];
  const float* Wv  = (const float*)d_in[3];
  const float* Wo  = (const float*)d_in[4];
  const float* Wg1 = (const float*)d_in[5];
  const float* Wg2 = (const float*)d_in[6];
  const float* lnw = (const float*)d_in[7];
  const float* lnb = (const float*)d_in[8];

  size_t off = 0;
  auto alloc = [&](size_t bytes) -> void* {
    void* p = (char*)d_ws + off;
    off += (bytes + 255) & ~(size_t)255;
    return p;
  };
  const size_t NE = (size_t)N_TOK * EMBED;   // 2M elems
  const size_t EE = (size_t)EMBED * EMBED;   // 1M elems

  bf16_t* x_bf   = (bf16_t*)alloc(NE * 2);
  bf16_t* wq_bf  = (bf16_t*)alloc(EE * 2);
  bf16_t* wk_bf  = (bf16_t*)alloc(EE * 2);
  bf16_t* wv_bf  = (bf16_t*)alloc(EE * 2);
  bf16_t* wo_bf  = (bf16_t*)alloc(EE * 2);
  bf16_t* wg1_bf = (bf16_t*)alloc((size_t)EMBED * HD * 2);
  bf16_t* wg2_bf = (bf16_t*)alloc((size_t)HD * EMBED * 2);
  bf16_t* q_bf   = (bf16_t*)alloc(NE * 2);
  bf16_t* k_bf   = (bf16_t*)alloc(NE * 2);
  bf16_t* v_bf   = (bf16_t*)alloc(NE * 2);
  bf16_t* g1_bf  = (bf16_t*)alloc((size_t)N_TOK * HD * 2);
  float*  attn_f = (float*)alloc(NE * 4);
  float*  gate_f = (float*)alloc(NE * 4);
  bf16_t* nrm_bf = (bf16_t*)alloc(NE * 2);

  auto cvt = [&](const float* s, bf16_t* d, size_t n) {
    cvt_f32_bf16<<<dim3((unsigned)((n + 255) / 256)), 256, 0, stream>>>(s, d, (int)n);
  };
  cvt(x,   x_bf,   NE);
  cvt(Wq,  wq_bf,  EE);
  cvt(Wk,  wk_bf,  EE);
  cvt(Wv,  wv_bf,  EE);
  cvt(Wo,  wo_bf,  EE);
  cvt(Wg1, wg1_bf, (size_t)EMBED * HD);
  cvt(Wg2, wg2_bf, (size_t)HD * EMBED);

  auto gemm = [&](const bf16_t* A, const bf16_t* B, float* Cf, bf16_t* Cb,
                  int M, int N, int K, int epi) {
    dim3 grid((N + BN - 1) / BN, (M + BM - 1) / BM);
    if (N % BN == 0)
      gemm_bf16_kernel<false><<<grid, 256, 0, stream>>>(A, B, Cf, Cb, M, N, K, epi);
    else
      gemm_bf16_kernel<true><<<grid, 256, 0, stream>>>(A, B, Cf, Cb, M, N, K, epi);
  };

  // q = silu(x@Wq), k = silu(x@Wk), v = x@Wv   (bf16 outputs)
  gemm(x_bf, wq_bf, nullptr, q_bf, N_TOK, EMBED, EMBED, 1);
  gemm(x_bf, wk_bf, nullptr, k_bf, N_TOK, EMBED, EMBED, 1);
  gemm(x_bf, wv_bf, nullptr, v_bf, N_TOK, EMBED, EMBED, 0);

  // RoPE in place on q,k
  rope_kernel<<<dim3(N_TOK * HEADS * 32 / 256), 256, 0, stream>>>(q_bf);
  rope_kernel<<<dim3(N_TOK * HEADS * 32 / 256), 256, 0, stream>>>(k_bf);

  // attention: (QK^T .* tril) V per head
  attn_kernel<<<dim3(HEADS, N_TOK / 64), 128, 0, stream>>>(q_bf, k_bf, v_bf, attn_f);

  // gate logits = (x@Wg1)@Wg2  (sigmoid applied in gate_ln)
  gemm(x_bf, wg1_bf, nullptr, g1_bf, N_TOK, HD, EMBED, 0);
  gemm(g1_bf, wg2_bf, gate_f, nullptr, N_TOK, EMBED, HD, 0);

  // out = LN(attn * sigmoid(gate)) -> bf16
  gate_ln_kernel<<<dim3(N_TOK), 256, 0, stream>>>(attn_f, gate_f, lnw, lnb, nrm_bf);

  // final: d_out = nrm @ Wo  (fp32 output)
  gemm(nrm_bf, wo_bf, (float*)d_out, nullptr, N_TOK, EMBED, EMBED, 0);
}
